// CentroidTripletLoss_56848187129916
// MI455X (gfx1250) — matrix operand order repaired and checked by reading, not compiled, not run on hardware
//
#include <hip/hip_runtime.h>
#include <hip/hip_bf16.h>

// CentroidTripletLoss on MI455X (gfx1250, wave32).
// Quadratic forms expanded so the 34 GFLOP lands in v_wmma_f32_16x16x32_bf16
// (split-precision bf16x3 emulation of fp32 GEMM).
//  - cov_inv pre-split to bf16 hi/lo, pre-swizzled to B-fragment layout.
//    hi and lo buffers are CONTIGUOUS (lo = hi + FRAGN) so the hot loop
//    addresses both from one pointer with constant 24-bit IOFFSETs.
//  - Each block converts its 16 A/E rows to bf16 hi/lo A-fragments ONCE into
//    LDS (64 KB, fragment layout); k-loop reads them as constant-offset
//    ds_load_b128 off a single loop-carried LDS address.
//  - WMMA issued over 4 independent accumulator chains (tile pairs x {A,E}).
//  - k-loop pinned to unroll 1: one iteration already exposes 48 WMMAs of ILP;
//    more unrolling pushes VGPRs past 256 and drags in s_set_vgpr_msb churn.
// All reductions deterministic (no global float atomics).

#define DIM  512
#define NCLS 128
#define TROWS 16
#define NSLICE 16
#define KBLKS (DIM / 32)                 // 16 K-blocks of 32
#define NTILE (DIM / 16)                 // 32 column tiles of 16
#define FRAGN (KBLKS * NTILE * 32 * 8)   // dwords per packed-M buffer (131072)
#define AFRAGD (KBLKS * 32 * 8)          // dwords per A-frag array (4096)
#define KSTRIDE (NTILE * 32 * 8)         // dwords per kblk in packed-M (8192)

typedef __attribute__((ext_vector_type(16))) __bf16 v16bf;
typedef __attribute__((ext_vector_type(8)))  float  v8f;
typedef __attribute__((ext_vector_type(8)))  unsigned int v8u;

__device__ __forceinline__ void split2(float x, __bf16& h, __bf16& l) {
    h = (__bf16)x;                 // RNE to bf16 (hi part)
    l = (__bf16)(x - (float)h);    // residual (lo part)
}

__device__ __forceinline__ unsigned int packbf(__bf16 lo16, __bf16 hi16) {
    return ((unsigned int)__builtin_bit_cast(unsigned short, hi16) << 16) |
           (unsigned int)__builtin_bit_cast(unsigned short, lo16);
}

// fragment K-offset pattern (16-bit A 16x32): lanes 0-15 -> K 0-15,
// lanes 16-31 -> K 16-31; dword v holds the (k, k+1) bf16 pair.
__device__ __forceinline__ int frag_ko(int v, int hfp) {
    return hfp * 16 + 2 * v;
}

// ---------------------------------------------------------------------------
// 0) Pre-split cov_inv into bf16 hi/lo, pre-swizzled into B-fragment layout:
//    dword index = ((kblk*NTILE + tile)*32 + lane)*8 + v
//    dword       = bf16(M[k][col]) | bf16(M[k+1][col]) << 16
//    with col = tile*16 + (lane&15), k = kblk*32 + (lane>>4)*16 + 2v
// ---------------------------------------------------------------------------
__global__ __launch_bounds__(256) void ctl_packM_kernel(
    const float* __restrict__ Mm, unsigned int* __restrict__ mpk /* hi | lo */)
{
    const int idx = blockIdx.x * 256 + threadIdx.x;
    if (idx >= FRAGN) return;
    const int v    = idx & 7;
    const int lane = (idx >> 3) & 31;
    const int t    = (idx >> 8) & (NTILE - 1);
    const int kblk = idx >> 13;
    const int col  = t * 16 + (lane & 15);
    const int k    = kblk * 32 + (lane >> 4) * 16 + 2 * v;
    const float m0 = Mm[(size_t)k * DIM + col];
    const float m1 = Mm[(size_t)(k + 1) * DIM + col];
    __bf16 h0, l0, h1, l1;
    split2(m0, h0, l0);
    split2(m1, h1, l1);
    mpk[idx]         = packbf(h0, h1);   // hi buffer
    mpk[idx + FRAGN] = packbf(l0, l1);   // lo buffer (contiguous)
}

// ---------------------------------------------------------------------------
// 1) Per-class partial sums over NSLICE row slices (deterministic, no atomics)
// ---------------------------------------------------------------------------
__global__ __launch_bounds__(256) void ctl_class_sum_kernel(
    const float* __restrict__ E_, const int* __restrict__ labels,
    float* __restrict__ spart, float* __restrict__ cpart, int B)
{
    const int c    = blockIdx.y;
    const int sl   = blockIdx.x;
    const int rows = (B + NSLICE - 1) / NSLICE;
    const int ibeg = sl * rows;
    const int iend = (ibeg + rows < B) ? (ibeg + rows) : B;
    const int tid  = threadIdx.x;   // 256 threads -> 2 dims each

    float a0 = 0.f, a1 = 0.f;
    int   cnt = 0;
    for (int i = ibeg; i < iend; ++i) {
        if (labels[i] == c) {       // uniform per iteration -> scalar compare
            const float* e = E_ + (size_t)i * DIM;
            a0 += e[tid];
            a1 += e[tid + 256];
            ++cnt;
        }
    }
    float* dst = spart + ((size_t)sl * NCLS + c) * DIM;
    dst[tid]       = a0;
    dst[tid + 256] = a1;
    if (tid == 0) cpart[sl * NCLS + c] = (float)cnt;
}

__global__ __launch_bounds__(256) void ctl_combine_kernel(
    const float* __restrict__ spart, const float* __restrict__ cpart,
    float* __restrict__ sums, float* __restrict__ cnts)
{
    const int idx = blockIdx.x * 256 + threadIdx.x;
    if (idx < NCLS * DIM) {
        float s = 0.f;
        #pragma unroll
        for (int sl = 0; sl < NSLICE; ++sl)
            s += spart[(size_t)sl * NCLS * DIM + idx];
        sums[idx] = s;
    }
    if (idx < NCLS) {
        float s = 0.f;
        #pragma unroll
        for (int sl = 0; sl < NSLICE; ++sl)
            s += cpart[sl * NCLS + idx];
        cnts[idx] = s;
    }
}

// ---------------------------------------------------------------------------
// 2) SM = sums @ M (67 MFLOP, trivial) and sMs[c] = SM[c].S[c]
// ---------------------------------------------------------------------------
__global__ __launch_bounds__(256) void ctl_smat_kernel(
    const float* __restrict__ sums, const float* __restrict__ Mm,
    float* __restrict__ SMo)
{
    const int c = blockIdx.y;
    const int n = blockIdx.x * 256 + threadIdx.x;
    const float* srow = sums + (size_t)c * DIM;
    float acc = 0.f;
    for (int k = 0; k < DIM; ++k)
        acc = fmaf(srow[k], Mm[(size_t)k * DIM + n], acc);
    SMo[(size_t)c * DIM + n] = acc;
}

__global__ __launch_bounds__(256) void ctl_sms_kernel(
    const float* __restrict__ SMo, const float* __restrict__ sums,
    float* __restrict__ sMs)
{
    const int c = blockIdx.x, tid = threadIdx.x;
    float p = 0.f;
    for (int d = tid; d < DIM; d += 256)
        p = fmaf(SMo[(size_t)c * DIM + d], sums[(size_t)c * DIM + d], p);
    for (int o = 16; o > 0; o >>= 1) p += __shfl_down(p, o);
    __shared__ float w[8];
    if ((tid & 31) == 0) w[tid >> 5] = p;
    __syncthreads();
    if (tid == 0) {
        float s = 0.f;
        #pragma unroll
        for (int j = 0; j < 8; ++j) s += w[j];
        sMs[c] = s;
    }
}

// ---------------------------------------------------------------------------
// 3) Main WMMA kernel. Block = 128 threads (4 waves); 16 rows; wave w owns
//    columns [w*128, w*128+128). A/E fragments staged once in LDS.
// ---------------------------------------------------------------------------
__global__ __launch_bounds__(128) void ctl_qform_wmma_kernel(
    const float* __restrict__ A_, const float* __restrict__ E_,
    const unsigned int* __restrict__ mpk /* hi | lo, contiguous */,
    const float* __restrict__ SMo,
    const int* __restrict__ labels, const int* __restrict__ neglab,
    float* __restrict__ aMa, float* __restrict__ aMe, float* __restrict__ eMe,
    float* __restrict__ aSMl, float* __restrict__ aSMn, float* __restrict__ eSMl,
    int B)
{
    // 64 KB LDS: 4 A-frag arrays [aHi|aLo|eHi|eLo][kblk][lane][8 dwords];
    // after the k-loop (sync) the front of the buffer is reused for reductions.
    __shared__ unsigned int shbuf[4 * AFRAGD];

    const int i0   = blockIdx.x * TROWS;
    const int lane = threadIdx.x & 31;
    const int wv   = threadIdx.x >> 5;     // wave 0..3
    const int n0w  = wv * 128;             // wave's column base

    // ---- one-time: convert 16 A rows + 16 E rows into LDS fragment layout
    for (int idx = threadIdx.x; idx < AFRAGD; idx += 128) {
        const int v   = idx & 7;
        const int ln  = (idx >> 3) & 31;
        const int kb  = idx >> 8;
        const int row = ln & 15;
        const int hfp = ln >> 4;
        const int k   = kb * 32 + frag_ko(v, hfp);
        const float* ar = A_ + (size_t)(i0 + row) * DIM + k;
        const float* er = E_ + (size_t)(i0 + row) * DIM + k;
        __bf16 h0, l0, h1, l1;
        split2(ar[0], h0, l0); split2(ar[1], h1, l1);
        shbuf[0 * AFRAGD + idx] = packbf(h0, h1);
        shbuf[1 * AFRAGD + idx] = packbf(l0, l1);
        split2(er[0], h0, l0); split2(er[1], h1, l1);
        shbuf[2 * AFRAGD + idx] = packbf(h0, h1);
        shbuf[3 * AFRAGD + idx] = packbf(l0, l1);
    }
    __syncthreads();

    v8f accA[8], accE[8];
    #pragma unroll
    for (int t = 0; t < 8; ++t)
        #pragma unroll
        for (int q = 0; q < 8; ++q) { accA[t][q] = 0.f; accE[t][q] = 0.f; }

    // loop-carried bases; everything else is a compile-time instruction offset
    const unsigned int* mb = mpk + ((size_t)(wv * 8) * 32 + lane) * 8;
    const unsigned int* sb = shbuf + lane * 8;

    #pragma unroll 1   // keep live set < 256 VGPRs (avoid s_set_vgpr_msb)
    for (int kblk = 0; kblk < KBLKS; ++kblk) {
        // prefetch next K-block of the packed-M fragments (L2-resident stream)
        if (kblk + 1 < KBLKS) {
            __builtin_prefetch(mb + KSTRIDE, 0, 1);
            __builtin_prefetch(mb + KSTRIDE + FRAGN, 0, 1);
        }

        // A/E fragments from LDS: 8x ds_load_b128, constant offsets off sb
        const v16bf aHi = __builtin_bit_cast(v16bf, *(const v8u*)(sb + 0 * AFRAGD));
        const v16bf aLo = __builtin_bit_cast(v16bf, *(const v8u*)(sb + 1 * AFRAGD));
        const v16bf eHi = __builtin_bit_cast(v16bf, *(const v8u*)(sb + 2 * AFRAGD));
        const v16bf eLo = __builtin_bit_cast(v16bf, *(const v8u*)(sb + 3 * AFRAGD));

        // tiles in pairs -> 4 independent accumulator chains between D-reuse
        #pragma unroll
        for (int tp = 0; tp < 4; ++tp) {
            const int t0 = 2 * tp, t1 = 2 * tp + 1;
            // tile stride = 256 dwords (1 KB); lo buffer at +FRAGN (512 KB):
            // all constant IOFFSETs off the single loop-carried pointer mb.
            const v16bf bHi0 = __builtin_bit_cast(v16bf, *(const v8u*)(mb + t0 * 256));
            const v16bf bLo0 = __builtin_bit_cast(v16bf, *(const v8u*)(mb + t0 * 256 + FRAGN));
            const v16bf bHi1 = __builtin_bit_cast(v16bf, *(const v8u*)(mb + t1 * 256));
            const v16bf bLo1 = __builtin_bit_cast(v16bf, *(const v8u*)(mb + t1 * 256 + FRAGN));
            // split-precision: hi*hi + hi*lo + lo*hi (lo*lo dropped); rotate
            // across 4 chains so same-D WMMAs are >=4 apart (hazard window).
            accA[t0] = __builtin_amdgcn_wmma_f32_16x16x32_bf16(false, aHi, false, bHi0, (short)0, accA[t0], false, false);
            accE[t0] = __builtin_amdgcn_wmma_f32_16x16x32_bf16(false, eHi, false, bHi0, (short)0, accE[t0], false, false);
            accA[t1] = __builtin_amdgcn_wmma_f32_16x16x32_bf16(false, aHi, false, bHi1, (short)0, accA[t1], false, false);
            accE[t1] = __builtin_amdgcn_wmma_f32_16x16x32_bf16(false, eHi, false, bHi1, (short)0, accE[t1], false, false);
            accA[t0] = __builtin_amdgcn_wmma_f32_16x16x32_bf16(false, aHi, false, bLo0, (short)0, accA[t0], false, false);
            accE[t0] = __builtin_amdgcn_wmma_f32_16x16x32_bf16(false, eHi, false, bLo0, (short)0, accE[t0], false, false);
            accA[t1] = __builtin_amdgcn_wmma_f32_16x16x32_bf16(false, aHi, false, bLo1, (short)0, accA[t1], false, false);
            accE[t1] = __builtin_amdgcn_wmma_f32_16x16x32_bf16(false, eHi, false, bLo1, (short)0, accE[t1], false, false);
            accA[t0] = __builtin_amdgcn_wmma_f32_16x16x32_bf16(false, aLo, false, bHi0, (short)0, accA[t0], false, false);
            accE[t0] = __builtin_amdgcn_wmma_f32_16x16x32_bf16(false, eLo, false, bHi0, (short)0, accE[t0], false, false);
            accA[t1] = __builtin_amdgcn_wmma_f32_16x16x32_bf16(false, aLo, false, bHi1, (short)0, accA[t1], false, false);
            accE[t1] = __builtin_amdgcn_wmma_f32_16x16x32_bf16(false, eLo, false, bHi1, (short)0, accE[t1], false, false);
        }
        mb += KSTRIDE;      // +32 KB: one 64-bit add per iteration
        sb += 32 * 8;       // +1 KB LDS: one 32-bit add per iteration
    }
    __syncthreads();   // fragments dead; reuse LDS front for reductions

    float* rQ = (float*)shbuf;             // [3][TROWS]
    float* rS = (float*)shbuf + 3 * TROWS; // [3][TROWS][8]

    // --- epilogue A: row-dot the Q strips against a/e (C/D layout: VGPR g is
    // row hf*8+g, column = lane&15 within tile) -> aMa, aMe, eMe via LDS.
    const int hf = lane >> 4;
    if (threadIdx.x < TROWS) {
        rQ[0 * TROWS + threadIdx.x] = 0.f;
        rQ[1 * TROWS + threadIdx.x] = 0.f;
        rQ[2 * TROWS + threadIdx.x] = 0.f;
    }
    __syncthreads();
    #pragma unroll
    for (int g = 0; g < 8; ++g) {
        const int row = hf * 8 + g;
        const float* ar = A_ + (size_t)(i0 + row) * DIM;
        const float* er = E_ + (size_t)(i0 + row) * DIM;
        float sAa = 0.f, sAe = 0.f, sEe = 0.f;
        #pragma unroll
        for (int t = 0; t < 8; ++t) {
            const int col = n0w + t * 16 + (lane & 15);
            const float av = ar[col], ev = er[col];
            sAa = fmaf(accA[t][g], av, sAa);
            sAe = fmaf(accA[t][g], ev, sAe);
            sEe = fmaf(accE[t][g], ev, sEe);
        }
        atomicAdd(&rQ[0 * TROWS + row], sAa);   // ds_add_f32, block-local
        atomicAdd(&rQ[1 * TROWS + row], sAe);
        atomicAdd(&rQ[2 * TROWS + row], sEe);
    }
    __syncthreads();

    // --- epilogue B: SM-row dot products (a.SM[l], a.SM[n], e.SM[l])
    {
        const int rr = threadIdx.x & 15;
        const int ch = threadIdx.x >> 4;         // 8 chunks of 64 columns
        const int li = labels[i0 + rr], ni = neglab[i0 + rr];
        const float* ar  = A_  + (size_t)(i0 + rr) * DIM;
        const float* er  = E_  + (size_t)(i0 + rr) * DIM;
        const float* sml = SMo + (size_t)li * DIM;
        const float* smn = SMo + (size_t)ni * DIM;
        float p0 = 0.f, p1 = 0.f, p2 = 0.f;
        for (int c = ch * 64; c < ch * 64 + 64; ++c) {
            const float av = ar[c], ev = er[c];
            p0 = fmaf(av, sml[c], p0);
            p1 = fmaf(av, smn[c], p1);
            p2 = fmaf(ev, sml[c], p2);
        }
        rS[(0 * TROWS + rr) * 8 + ch] = p0;
        rS[(1 * TROWS + rr) * 8 + ch] = p1;
        rS[(2 * TROWS + rr) * 8 + ch] = p2;
    }
    __syncthreads();

    if (threadIdx.x < TROWS) {
        const int rr = threadIdx.x;
        float s0 = 0.f, s1 = 0.f, s2 = 0.f;
        #pragma unroll
        for (int ch = 0; ch < 8; ++ch) {
            s0 += rS[(0 * TROWS + rr) * 8 + ch];
            s1 += rS[(1 * TROWS + rr) * 8 + ch];
            s2 += rS[(2 * TROWS + rr) * 8 + ch];
        }
        const int gi = i0 + rr;
        aMa[gi]  = rQ[0 * TROWS + rr];
        aMe[gi]  = rQ[1 * TROWS + rr];
        eMe[gi]  = rQ[2 * TROWS + rr];
        aSMl[gi] = s0;
        aSMn[gi] = s1;
        eSMl[gi] = s2;
    }
}

// ---------------------------------------------------------------------------
// 4) Per-sample loss -> per-block partials (deterministic), then final reduce
// ---------------------------------------------------------------------------
__global__ __launch_bounds__(256) void ctl_loss_kernel(
    const float* __restrict__ aMa, const float* __restrict__ aMe,
    const float* __restrict__ eMe, const float* __restrict__ aSMl,
    const float* __restrict__ aSMn, const float* __restrict__ eSMl,
    const float* __restrict__ cnts, const float* __restrict__ sMs,
    const int* __restrict__ labels, const int* __restrict__ neglab,
    float* __restrict__ lpart, int B)
{
    const int i = blockIdx.x * 256 + threadIdx.x;
    float v = 0.f;
    if (i < B) {
        const int l = labels[i], n = neglab[i];
        const float al = 1.f / fmaxf(cnts[l] - 1.f, 1.f);
        const float be = 1.f / fmaxf(cnts[n], 1.f);
        // d2_pos = aMa - 2a*aSMl + 2a*aMe + a^2*(sMs_l + eMe - 2*eSMl)
        const float d2p = aMa[i] - 2.f * al * aSMl[i] + 2.f * al * aMe[i]
                        + al * al * (sMs[l] + eMe[i] - 2.f * eSMl[i]);
        const float d2n = aMa[i] - 2.f * be * aSMn[i] + be * be * sMs[n];
        const float pd = sqrtf(fmaxf(d2p, 0.f) + 1e-12f);
        const float nd = sqrtf(fmaxf(d2n, 0.f) + 1e-12f);
        v = fmaxf(pd - nd + 0.1f, 0.f);
    }
    for (int o = 16; o > 0; o >>= 1) v += __shfl_down(v, o);
    __shared__ float w[8];
    if ((threadIdx.x & 31) == 0) w[threadIdx.x >> 5] = v;
    __syncthreads();
    if (threadIdx.x == 0) {
        float s = 0.f;
        #pragma unroll
        for (int j = 0; j < 8; ++j) s += w[j];
        lpart[blockIdx.x] = s;
    }
}

__global__ __launch_bounds__(256) void ctl_final_kernel(
    const float* __restrict__ lpart, int nb, float* __restrict__ out, float invB)
{
    float v = 0.f;
    for (int i = threadIdx.x; i < nb; i += 256) v += lpart[i];
    for (int o = 16; o > 0; o >>= 1) v += __shfl_down(v, o);
    __shared__ float w[8];
    if ((threadIdx.x & 31) == 0) w[threadIdx.x >> 5] = v;
    __syncthreads();
    if (threadIdx.x == 0) {
        float s = 0.f;
        #pragma unroll
        for (int j = 0; j < 8; ++j) s += w[j];
        out[0] = s * invB;
    }
}

// ---------------------------------------------------------------------------
extern "C" void kernel_launch(void* const* d_in, const int* in_sizes, int n_in,
                              void* d_out, int out_size, void* d_ws, size_t ws_size,
                              hipStream_t stream) {
    const float* anchor = (const float*)d_in[0];
    const float* emb    = (const float*)d_in[1];
    const float* cov    = (const float*)d_in[2];
    const int*   labels = (const int*)d_in[3];
    const int*   negl   = (const int*)d_in[4];
    const int B = in_sizes[3];
    float* out = (float*)d_out;

    // workspace layout (floats)
    float* ws    = (float*)d_ws;
    unsigned int* mpk = (unsigned int*)ws;               // 2*FRAGN dwords (hi|lo)
    float* spart = ws + 2 * FRAGN;                       // NSLICE*NCLS*DIM
    float* cpart = spart + (size_t)NSLICE * NCLS * DIM;  // NSLICE*NCLS
    float* sums  = cpart + NSLICE * NCLS;                // NCLS*DIM
    float* cnts  = sums + NCLS * DIM;                    // NCLS
    float* SMo   = cnts + NCLS;                          // NCLS*DIM
    float* sMs   = SMo + NCLS * DIM;                     // NCLS
    float* aMa   = sMs + NCLS;                           // B each, x6
    float* aMe   = aMa + B;
    float* eMe   = aMe + B;
    float* aSMl  = eMe + B;
    float* aSMn  = aSMl + B;
    float* eSMl  = aSMn + B;
    float* lpart = eSMl + B;                             // nb

    const int nb = (B + 255) / 256;

    ctl_packM_kernel<<<(FRAGN + 255) / 256, 256, 0, stream>>>(cov, mpk);
    ctl_class_sum_kernel<<<dim3(NSLICE, NCLS), 256, 0, stream>>>(emb, labels, spart, cpart, B);
    ctl_combine_kernel<<<(NCLS * DIM + 255) / 256, 256, 0, stream>>>(spart, cpart, sums, cnts);
    ctl_smat_kernel<<<dim3(DIM / 256, NCLS), 256, 0, stream>>>(sums, cov, SMo);
    ctl_sms_kernel<<<NCLS, 256, 0, stream>>>(SMo, sums, sMs);
    ctl_qform_wmma_kernel<<<B / TROWS, 128, 0, stream>>>(anchor, emb, mpk, SMo, labels, negl,
                                                         aMa, aMe, eMe, aSMl, aSMn, eSMl, B);
    ctl_loss_kernel<<<nb, 256, 0, stream>>>(aMa, aMe, eMe, aSMl, aSMn, eSMl,
                                            cnts, sMs, labels, negl, lpart, B);
    ctl_final_kernel<<<1, 256, 0, stream>>>(lpart, nb, out, 1.f / (float)B);
}
